// FeedForward_77043123355776
// MI455X (gfx1250) — 8-bit matrix operand order repaired and checked by reading, compiled, not run on hardware
//
#include <hip/hip_runtime.h>
#include <hip/hip_bf16.h>
#include <math.h>


typedef int v8i __attribute__((ext_vector_type(8)));
typedef signed char s8;

#define DIM     4096
#define HIDDEN  11008
#define MTOK    4096
#define NW      ((size_t)HIDDEN * (size_t)DIM)
#define EPSQ    1e-5f

static const size_t OFF_SCALE = 0;
static const size_t OFF_PART  = 256;
static const size_t OFF_QW1   = 16384;
static const size_t OFF_QW3   = OFF_QW1 + NW;
static const size_t OFF_QW2   = OFF_QW3 + NW;
static const size_t OFF_QX    = OFF_QW2 + NW;
static const size_t OFF_QH    = OFF_QX  + (size_t)MTOK*DIM;
static const size_t OFF_DQX   = OFF_QH  + (size_t)MTOK*HIDDEN;
static const size_t OFF_DQH   = OFF_DQX + (size_t)MTOK*4;
static const size_t OFF_H     = OFF_DQH + (size_t)MTOK*4;

__global__ __launch_bounds__(256)
void k_partial_abs(const float* __restrict__ w, size_t n4, float* __restrict__ part)
{
    __shared__ float sm[256];
    const int tid = threadIdx.x;
    float s = 0.f;
    for (size_t i = (size_t)blockIdx.x * 256 + tid; i < n4; i += (size_t)1024 * 256) {
        float4 v = ((const float4*)w)[i];
        s += fabsf(v.x) + fabsf(v.y) + fabsf(v.z) + fabsf(v.w);
    }
    sm[tid] = s;
    __syncthreads();
    for (int off = 128; off > 0; off >>= 1) {
        if (tid < off) sm[tid] += sm[tid + off];
        __syncthreads();
    }
    if (tid == 0) part[blockIdx.x] = sm[0];
}

__global__ __launch_bounds__(256)
void k_finalize(const float* __restrict__ part, float* __restrict__ scales, float n)
{
    __shared__ float sm[256];
    const int tid = threadIdx.x;
    for (int t = 0; t < 3; ++t) {
        float s = 0.f;
        for (int i = tid; i < 1024; i += 256) s += part[t * 1024 + i];
        sm[tid] = s;
        __syncthreads();
        for (int off = 128; off > 0; off >>= 1) {
            if (tid < off) sm[tid] += sm[tid + off];
            __syncthreads();
        }
        if (tid == 0) scales[t] = fmaxf(sm[0] / n, EPSQ);
        __syncthreads();
    }
}

__global__ __launch_bounds__(256)
void k_quant_w(const float* __restrict__ w, s8* __restrict__ q,
               size_t n4, const float* __restrict__ sp)
{
    const size_t i = (size_t)blockIdx.x * 256 + threadIdx.x;
    if (i >= n4) return;
    const float inv_s = 1.0f / sp[0];
    float4 v = ((const float4*)w)[i];
    int b0 = (int)fminf(fmaxf(rintf(v.x * inv_s), -1.f), 1.f);
    int b1 = (int)fminf(fmaxf(rintf(v.y * inv_s), -1.f), 1.f);
    int b2 = (int)fminf(fmaxf(rintf(v.z * inv_s), -1.f), 1.f);
    int b3 = (int)fminf(fmaxf(rintf(v.w * inv_s), -1.f), 1.f);
    unsigned p = (unsigned)(b0 & 0xff) | ((unsigned)(b1 & 0xff) << 8) |
                 ((unsigned)(b2 & 0xff) << 16) | ((unsigned)(b3 & 0xff) << 24);
    ((unsigned*)q)[i] = p;
}

__global__ __launch_bounds__(256)
void k_act_quant(const float* __restrict__ x, s8* __restrict__ q,
                 float* __restrict__ dq, int rowlen)
{
    extern __shared__ float rowbuf[];
    __shared__ float sm[256];
    __shared__ float s_scale;
    const int row = blockIdx.x;
    const int tid = threadIdx.x;
    const int n4 = rowlen >> 2;
    const size_t base4 = (size_t)row * n4;

    float amax = 0.f;
    for (int i = tid; i < n4; i += 256) {
        float4 v = ((const float4*)x)[base4 + i];
        ((float4*)rowbuf)[i] = v;
        amax = fmaxf(amax, fmaxf(fmaxf(fabsf(v.x), fabsf(v.y)),
                                 fmaxf(fabsf(v.z), fabsf(v.w))));
    }
    sm[tid] = amax;
    __syncthreads();
    for (int off = 128; off > 0; off >>= 1) {
        if (tid < off) sm[tid] = fmaxf(sm[tid], sm[tid + off]);
        __syncthreads();
    }
    if (tid == 0) {
        float a = fmaxf(sm[0], EPSQ);
        s_scale = 127.0f / a;
        dq[row] = a / 127.0f;
    }
    __syncthreads();
    const float sc = s_scale;
    for (int i = tid; i < n4; i += 256) {
        float4 v = ((const float4*)rowbuf)[i];
        int b0 = (int)fminf(fmaxf(rintf(v.x * sc), -128.f), 127.f);
        int b1 = (int)fminf(fmaxf(rintf(v.y * sc), -128.f), 127.f);
        int b2 = (int)fminf(fmaxf(rintf(v.z * sc), -128.f), 127.f);
        int b3 = (int)fminf(fmaxf(rintf(v.w * sc), -128.f), 127.f);
        unsigned p = (unsigned)(b0 & 0xff) | ((unsigned)(b1 & 0xff) << 8) |
                     ((unsigned)(b2 & 0xff) << 16) | ((unsigned)(b3 & 0xff) << 24);
        ((unsigned*)q)[base4 + i] = p;
    }
}

template <bool FUSED>
__global__ __launch_bounds__(256)
void k_gemm_i8(const s8* __restrict__ qa, const float* __restrict__ dqa,
               const s8* __restrict__ qb1, const s8* __restrict__ qb3,
               const float* __restrict__ sb1p, const float* __restrict__ sb3p,
               float* __restrict__ out, int N, int K)
{
    __shared__ __align__(16) s8 lA [2][128 * 64];
    __shared__ __align__(16) s8 lB1[2][64 * 64];
    __shared__ __align__(16) s8 lB3[2][64 * 64];

    const int t      = threadIdx.x;
    const int m_base = blockIdx.y * 128;
    const int n_base = blockIdx.x * 64;

    const int lane = t & 31;
    const int wave = t >> 5;
    const int wm   = wave >> 1;
    const int wn   = wave & 1;
    const int l16  = lane & 15;
    const int hi   = lane >> 4;
    const int m_off = wm * 32;
    const int n_off = wn * 32;

    v8i acc1[2][2] = {};
    v8i acc3[2][2] = {};

    const int arow = t >> 2;
    const int acol = (t & 3) << 4;
    const int brow = t >> 2;
    const int bcol = (t & 3) << 4;

    const s8* gA0 = qa  + (size_t)(m_base + arow)      * K + acol;
    const s8* gA1 = qa  + (size_t)(m_base + arow + 64) * K + acol;
    const s8* gB1 = qb1 + (size_t)(n_base + brow)      * K + bcol;
    const s8* gB3 = qb3 + (size_t)(n_base + brow)      * K + bcol;

    int4 rA0, rA1, rB1, rB3;
    auto load_tile = [&](int k0) {
        rA0 = *(const int4*)(gA0 + k0);
        rA1 = *(const int4*)(gA1 + k0);
        rB1 = *(const int4*)(gB1 + k0);
        if (FUSED) rB3 = *(const int4*)(gB3 + k0);
    };
    auto store_tile = [&](int buf) {
        *(int4*)(lA[buf]  + arow * 64 + acol)        = rA0;
        *(int4*)(lA[buf]  + (arow + 64) * 64 + acol) = rA1;
        *(int4*)(lB1[buf] + brow * 64 + bcol)        = rB1;
        if (FUSED) *(int4*)(lB3[buf] + brow * 64 + bcol) = rB3;
    };

    const int ksteps = K >> 6;

    load_tile(0);
    store_tile(0);

    for (int kk = 0; kk < ksteps; ++kk) {
        __syncthreads();
        const int cur = kk & 1;
        const int nxt = cur ^ 1;

        if (kk + 1 < ksteps) load_tile((kk + 1) << 6);
        if (kk + 2 < ksteps) {
            __builtin_prefetch(gA0 + ((kk + 2) << 6), 0, 1);
            __builtin_prefetch(gB1 + ((kk + 2) << 6), 0, 1);
            if (FUSED) __builtin_prefetch(gB3 + ((kk + 2) << 6), 0, 1);
        }

        v8i a[2];
#pragma unroll
        for (int mt = 0; mt < 2; ++mt) {
            const unsigned* ar = (const unsigned*)(lA[cur] + (m_off + mt * 16 + l16) * 64);
#pragma unroll
            for (int j = 0; j < 4; ++j) {
                a[mt][2 * j]     = (int)ar[2 * j + 4 * hi + 4 * (j >> 1)];
                a[mt][2 * j + 1] = (int)ar[2 * j + 4 * hi + 4 * (j >> 1) + 1];
            }
        }

        v8i b1[2], b3[2];
#pragma unroll
        for (int nt = 0; nt < 2; ++nt) {
            const s8* p = lB1[cur] + (n_off + nt * 16 + l16) * 64 + hi * 16;
            int4 lo = *(const int4*)(p);
            int4 hi4 = *(const int4*)(p + 32);
            b1[nt][0] = lo.x;  b1[nt][1] = lo.y;  b1[nt][2] = lo.z;  b1[nt][3] = lo.w;
            b1[nt][4] = hi4.x; b1[nt][5] = hi4.y; b1[nt][6] = hi4.z; b1[nt][7] = hi4.w;
            if (FUSED) {
                const s8* p3 = lB3[cur] + (n_off + nt * 16 + l16) * 64 + hi * 16;
                int4 lo3 = *(const int4*)(p3);
                int4 hi3 = *(const int4*)(p3 + 32);
                b3[nt][0] = lo3.x;  b3[nt][1] = lo3.y;  b3[nt][2] = lo3.z;  b3[nt][3] = lo3.w;
                b3[nt][4] = hi3.x;  b3[nt][5] = hi3.y;  b3[nt][6] = hi3.z;  b3[nt][7] = hi3.w;
            }
        }

#pragma unroll
        for (int mt = 0; mt < 2; ++mt)
#pragma unroll
            for (int nt = 0; nt < 2; ++nt) {
                acc1[mt][nt] = __builtin_amdgcn_wmma_i32_16x16x64_iu8(
                    true, a[mt], true, b1[nt], acc1[mt][nt], false, false);
                if (FUSED)
                    acc3[mt][nt] = __builtin_amdgcn_wmma_i32_16x16x64_iu8(
                        true, a[mt], true, b3[nt], acc3[mt][nt], false, false);
            }

        if (kk + 1 < ksteps) store_tile(nxt);
    }

    const float sb1 = sb1p[0];
    const float sb3 = FUSED ? sb3p[0] : 0.0f;
#pragma unroll
    for (int mt = 0; mt < 2; ++mt) {
#pragma unroll
        for (int r = 0; r < 8; ++r) {
            const int m = m_base + m_off + mt * 16 + hi * 8 + r;
            const float dq = dqa[m];
#pragma unroll
            for (int nt = 0; nt < 2; ++nt) {
                const int n = n_base + n_off + nt * 16 + l16;
                float v1 = (float)acc1[mt][nt][r] * sb1 * dq;
                float res;
                if (FUSED) {
                    float v3 = (float)acc3[mt][nt][r] * sb3 * dq;
                    res = (v1 / (1.0f + __expf(-v1))) * v3;
                } else {
                    res = v1;
                }
                out[(size_t)m * N + n] = res;
            }
        }
    }
}

extern "C" void kernel_launch(void* const* d_in, const int* in_sizes, int n_in,
                              void* d_out, int out_size, void* d_ws, size_t ws_size,
                              hipStream_t stream)
{
    const float* x  = (const float*)d_in[0];
    const float* w1 = (const float*)d_in[1];
    const float* w2 = (const float*)d_in[2];
    const float* w3 = (const float*)d_in[3];

    char* ws = (char*)d_ws;
    float* scales = (float*)(ws + OFF_SCALE);
    float* part   = (float*)(ws + OFF_PART);
    s8*    qw1    = (s8*)(ws + OFF_QW1);
    s8*    qw3    = (s8*)(ws + OFF_QW3);
    s8*    qw2    = (s8*)(ws + OFF_QW2);
    s8*    qx     = (s8*)(ws + OFF_QX);
    s8*    qh     = (s8*)(ws + OFF_QH);
    float* dqx    = (float*)(ws + OFF_DQX);
    float* dqh    = (float*)(ws + OFF_DQH);
    float* h      = (float*)(ws + OFF_H);
    float* outp   = (float*)d_out;

    const size_t n4 = NW / 4;
    const int qgrid = (int)((n4 + 255) / 256);

    k_partial_abs<<<1024, 256, 0, stream>>>(w1, n4, part + 0 * 1024);
    k_partial_abs<<<1024, 256, 0, stream>>>(w2, n4, part + 1 * 1024);
    k_partial_abs<<<1024, 256, 0, stream>>>(w3, n4, part + 2 * 1024);
    k_finalize<<<1, 256, 0, stream>>>(part, scales, (float)NW);

    k_quant_w<<<qgrid, 256, 0, stream>>>(w1, qw1, n4, scales + 0);
    k_quant_w<<<qgrid, 256, 0, stream>>>(w2, qw2, n4, scales + 1);
    k_quant_w<<<qgrid, 256, 0, stream>>>(w3, qw3, n4, scales + 2);

    k_act_quant<<<MTOK, 256, DIM * 4, stream>>>(x, qx, dqx, DIM);

    k_gemm_i8<true><<<dim3(HIDDEN / 64, MTOK / 128), 256, 0, stream>>>(
        qx, dqx, qw1, qw3, scales + 0, scales + 2, h, HIDDEN, DIM);

    k_act_quant<<<MTOK, 256, HIDDEN * 4, stream>>>(h, qh, dqh, HIDDEN);
    k_gemm_i8<false><<<dim3(DIM / 64, MTOK / 128), 256, 0, stream>>>(
        qh, dqh, qw2, qw2, scales + 1, scales + 1, outp, DIM, HIDDEN);

    (void)in_sizes; (void)n_in; (void)out_size; (void)ws_size;
}
